// MultiModalMambaKANDecoder_19164144075112
// MI455X (gfx1250) — matrix operand order repaired and checked by reading, not compiled, not run on hardware
//
#include <hip/hip_runtime.h>
#include <hip/hip_bf16.h>

// ---- model dims ----
#define B_       8
#define LP_      1024
#define CIN_     64
#define AUX_     32
#define COUT_    64
#define OUT_LEN_ 512
#define DM_      1024
#define DI_      2048
#define NS_      16
#define PH_      64
#define NH_      32
#define NL_      4
#define PATCH_   8
#define KPRE_    5
#define KCONV_   4
#define GG_      8
#define KH_      2048
#define L_       128                    // (LP/PATCH), already multiple of CHUNK
#define DIN_     (2*DI_ + 2*NS_ + NH_)  // 4160
#define CONV_DIM_ (DI_ + 2*NS_)         // 2080
#define MROWS_   (B_*L_)                // 1024

typedef __attribute__((ext_vector_type(16))) __bf16 v16bf_t;
typedef __attribute__((ext_vector_type(8)))  float  v8f_t;

// Native f32 -> bf16 (gfx1250 has HW bf16 converts; RNE)
__device__ __forceinline__ __bf16 f2bf(float f) { return (__bf16)f; }

__device__ __forceinline__ float silu_f(float v) { return v / (1.f + __expf(-v)); }

// ---------------------------------------------------------------------------
// Generic WMMA GEMM: D[m,n] = act( sum_k A[m,k]*W[n,k] + bias[n] (+ resid[m,n]) )
// A: MxK f32 row-major (activations), W: NxK f32 row-major (torch-style weight)
// Converted to bf16 in registers; v_wmma_f32_16x16x32_bf16, f32 accumulate.
// grid(M/16, ceil(N/256)), block(128) = 4 waves, each wave a 16x64 tile.
// act: 0=none, 1=silu
// ---------------------------------------------------------------------------
__global__ void __launch_bounds__(128)
wmma_gemm_f32bf16(const float* __restrict__ A, const float* __restrict__ W,
                  const float* __restrict__ bias, const float* resid,
                  float* D, int M, int N, int K, int act)
{
  const int wave = threadIdx.x >> 5;
  const int lane = threadIdx.x & 31;
  const int half = lane >> 4;
  const int l16  = lane & 15;
  const int m0 = blockIdx.x * 16;
  const int n0 = blockIdx.y * 256 + wave * 64;
  if (n0 >= N) return;                       // wave-uniform; EXEC stays all-1s

  const float* Arow = A + (size_t)(m0 + l16) * K;
  const float* W0 = W + (size_t)(n0 +  0 + l16) * K;
  const float* W1 = W + (size_t)(n0 + 16 + l16) * K;
  const float* W2 = W + (size_t)(n0 + 32 + l16) * K;
  const float* W3 = W + (size_t)(n0 + 48 + l16) * K;

  v8f_t acc0 = {}, acc1 = {}, acc2 = {}, acc3 = {};

  for (int k0 = 0; k0 < K; k0 += 32) {
    if (k0 + 256 < K) {                      // stream-ahead prefetch (global_prefetch_b8)
      __builtin_prefetch(Arow + k0 + 256, 0, 0);
      __builtin_prefetch(W0 + k0 + 256, 0, 0);
      __builtin_prefetch(W2 + k0 + 256, 0, 0);
    }
    v16bf_t a, b0, b1, b2, b3;
    // A 16x32 bf16 fragment: lanes 0-15 & 16-31 both hold M=l16;
    // k = (j>>2)*16 + half*8 + (j&3)*2
    #pragma unroll
    for (int j = 0; j < 8; ++j) {
      int k = k0 + ((j >> 2) << 4) + (half << 3) + ((j & 3) << 1);
      float2 av = *(const float2*)(Arow + k);
      a[2*j] = f2bf(av.x); a[2*j+1] = f2bf(av.y);
    }
    // B 32x16 bf16 fragment: N=l16; same k order as A;
    // B[k,n] = W[n,k]  ->  k = (j>>2)*16 + half*8 + (j&3)*2
    #pragma unroll
    for (int j = 0; j < 8; ++j) {
      int k = k0 + ((j >> 2) << 4) + (half << 3) + ((j & 3) << 1);
      float2 w0 = *(const float2*)(W0 + k);
      float2 w1 = *(const float2*)(W1 + k);
      float2 w2 = *(const float2*)(W2 + k);
      float2 w3 = *(const float2*)(W3 + k);
      b0[2*j] = f2bf(w0.x); b0[2*j+1] = f2bf(w0.y);
      b1[2*j] = f2bf(w1.x); b1[2*j+1] = f2bf(w1.y);
      b2[2*j] = f2bf(w2.x); b2[2*j+1] = f2bf(w2.y);
      b3[2*j] = f2bf(w3.x); b3[2*j+1] = f2bf(w3.y);
    }
    acc0 = __builtin_amdgcn_wmma_f32_16x16x32_bf16(false, a, false, b0, (short)0, acc0, false, false);
    acc1 = __builtin_amdgcn_wmma_f32_16x16x32_bf16(false, a, false, b1, (short)0, acc1, false, false);
    acc2 = __builtin_amdgcn_wmma_f32_16x16x32_bf16(false, a, false, b2, (short)0, acc2, false, false);
    acc3 = __builtin_amdgcn_wmma_f32_16x16x32_bf16(false, a, false, b3, (short)0, acc3, false, false);
  }

  // D fragment: VGPR r holds row m0 + half*8 + r, col n0 + t*16 + l16
  #pragma unroll
  for (int t = 0; t < 4; ++t) {
    v8f_t acc = (t == 0) ? acc0 : (t == 1) ? acc1 : (t == 2) ? acc2 : acc3;
    int n = n0 + t * 16 + l16;
    float bv = bias ? bias[n] : 0.f;
    #pragma unroll
    for (int r = 0; r < 8; ++r) {
      int m = m0 + (half << 3) + r;
      size_t idx = (size_t)m * N + n;
      float v = acc[r] + bv;
      if (resid) v += resid[idx];
      if (act == 1) v = silu_f(v);
      D[idx] = v;
    }
  }
}

// ---- im2col for pre-conv (kernel 5, pad 2): A[(b*LP+l), c*5+kk] ----
__global__ void preconv_im2col(const float* __restrict__ xp, float* __restrict__ A) {
  int i = blockIdx.x * blockDim.x + threadIdx.x;
  const int KC = CIN_ * KPRE_;
  if (i >= B_ * LP_ * KC) return;
  int col = i % KC, row = i / KC;
  int c = col / KPRE_, kk = col % KPRE_;
  int b = row / LP_, l = row % LP_;
  int ls = l + kk - KPRE_ / 2;
  float v = 0.f;
  if (ls >= 0 && ls < LP_) v = xp[((size_t)b * LP_ + ls) * CIN_ + c];
  A[(size_t)row * KC + col] = v;
}

// ---- im2col for patch conv (kernel 8, stride 8, no overlap -> reshape) ----
// xc layout: (B, LP, DM) from pre-conv GEMM.  A[(b*L+t), c*8+k] = xc[b, t*8+k, c]
__global__ void patch_im2col(const float* __restrict__ xc, float* __restrict__ A) {
  int i = blockIdx.x * blockDim.x + threadIdx.x;
  const int KC = DM_ * PATCH_;
  if (i >= MROWS_ * KC) return;
  int col = i % KC, row = i / KC;
  int c = col / PATCH_, k = col % PATCH_;
  int b = row / L_, t = row % L_;
  A[(size_t)row * KC + col] = xc[((size_t)b * LP_ + t * PATCH_ + k) * DM_ + c];
}

// ---- h_f = x ; h_b = x reversed along L ----
__global__ void init_dirs(const float* __restrict__ x, float* __restrict__ hf,
                          float* __restrict__ hb) {
  int i = blockIdx.x * blockDim.x + threadIdx.x;
  if (i >= MROWS_ * DM_) return;
  int d = i % DM_, bt = i / DM_;
  int b = bt / L_, t = bt % L_;
  hf[i] = x[i];
  hb[i] = x[((size_t)(b * L_ + (L_ - 1 - t))) * DM_ + d];
}

// ---- RMSNorm over DM, one block per (b,t) row ----
__global__ void __launch_bounds__(256)
rmsnorm_k(const float* __restrict__ h, const float* __restrict__ w, float* __restrict__ out) {
  int row = blockIdx.x;
  const float* x = h + (size_t)row * DM_;
  __shared__ float red[256];
  float ss = 0.f;
  for (int d = threadIdx.x; d < DM_; d += 256) { float v = x[d]; ss += v * v; }
  red[threadIdx.x] = ss; __syncthreads();
  for (int s = 128; s > 0; s >>= 1) {
    if (threadIdx.x < s) red[threadIdx.x] += red[threadIdx.x + s];
    __syncthreads();
  }
  float scale = rsqrtf(red[0] / DM_ + 1e-5f);
  for (int d = threadIdx.x; d < DM_; d += 256)
    out[(size_t)row * DM_ + d] = x[d] * scale * w[d];
}

// ---- dt = softplus(raw + dt_bias), dA = exp(-exp(A_log)*dt) ----
__global__ void dtda_k(const float* __restrict__ zx, const float* __restrict__ dt_bias,
                       const float* __restrict__ A_log, float* __restrict__ dt,
                       float* __restrict__ dA) {
  int i = blockIdx.x * blockDim.x + threadIdx.x;
  if (i >= MROWS_ * NH_) return;
  int h = i % NH_, bt = i / NH_;
  float raw = zx[(size_t)bt * DIN_ + DI_ + CONV_DIM_ + h] + dt_bias[h];
  float dtv = (raw > 20.f) ? raw : log1pf(__expf(raw));
  dt[i] = dtv;
  dA[i] = __expf(-__expf(A_log[h]) * dtv);
}

// ---- depthwise causal conv (K=4, pad 3 left) + silu over xBC channels ----
__global__ void dwconv_k(const float* __restrict__ zx, const float* __restrict__ cw,
                         const float* __restrict__ cb, float* __restrict__ out) {
  int i = blockIdx.x * blockDim.x + threadIdx.x;
  if (i >= MROWS_ * CONV_DIM_) return;
  int c = i % CONV_DIM_, bt = i / CONV_DIM_;
  int b = bt / L_, t = bt % L_;
  float acc = cb[c];
  #pragma unroll
  for (int k = 0; k < KCONV_; ++k) {
    int tt = t + k - (KCONV_ - 1);
    if (tt >= 0)
      acc += zx[((size_t)(b * L_ + tt)) * DIN_ + DI_ + c] * cw[c * KCONV_ + k];
  }
  out[(size_t)bt * CONV_DIM_ + c] = silu_f(acc);
}

// ---- selective-state scan: one block per (b,h), thread p owns NS states ----
__global__ void __launch_bounds__(64)
scan_k(const float* __restrict__ zx, const float* __restrict__ xbc,
       const float* __restrict__ dt, const float* __restrict__ dA,
       const float* __restrict__ Dh, float* __restrict__ ys) {
  int b = blockIdx.x / NH_, h = blockIdx.x % NH_;
  int p = threadIdx.x;
  __shared__ float sB[NS_], sC[NS_], sMisc[2];
  float s[NS_];
  #pragma unroll
  for (int n = 0; n < NS_; ++n) s[n] = 0.f;
  float dv = Dh[h];
  for (int t = 0; t < L_; ++t) {
    int bt = b * L_ + t;
    if (p < NS_)            sB[p]        = xbc[(size_t)bt * CONV_DIM_ + DI_ + p];
    else if (p < 2 * NS_)   sC[p - NS_]  = xbc[(size_t)bt * CONV_DIM_ + DI_ + NS_ + (p - NS_)];
    else if (p == 2 * NS_) { sMisc[0] = dA[bt * NH_ + h]; sMisc[1] = dt[bt * NH_ + h]; }
    __syncthreads();
    float x   = xbc[(size_t)bt * CONV_DIM_ + h * PH_ + p];
    float da  = sMisc[0];
    float xdt = x * sMisc[1];
    float y = 0.f;
    #pragma unroll
    for (int n = 0; n < NS_; ++n) { s[n] = s[n] * da + xdt * sB[n]; y += s[n] * sC[n]; }
    y += x * dv;
    float z = zx[(size_t)bt * DIN_ + h * PH_ + p];
    ys[(size_t)bt * DI_ + h * PH_ + p] = y * silu_f(z);
    __syncthreads();
  }
}

// ---- ctx = [mean_t h_f, mean_t h_b] into fused[:, 0:2048] ----
__global__ void mean_ctx_k(const float* __restrict__ hf, const float* __restrict__ hb,
                           float* __restrict__ fused) {
  int i = blockIdx.x * blockDim.x + threadIdx.x;
  if (i >= 2 * B_ * DM_) return;
  int d = i % DM_; int rest = i / DM_;
  int b = rest % B_; int which = rest / B_;
  const float* src = which ? hb : hf;
  float s = 0.f;
  for (int t = 0; t < L_; ++t) s += src[((size_t)(b * L_ + t)) * DM_ + d];
  fused[(size_t)b * (3 * DM_) + which * DM_ + d] = s / (float)L_;
}

// ---- KAN Gaussian basis: basis[b, f*GG+g] = exp(-(x-grid_g)^2/(2 bw_f^2)) ----
__global__ void basis_k(const float* __restrict__ x, const float* __restrict__ log_bw,
                        float* __restrict__ basis, int F, int ldx) {
  int i = blockIdx.x * blockDim.x + threadIdx.x;
  if (i >= B_ * F) return;
  int f = i % F, b = i / F;
  float bw = __expf(log_bw[f]) + 1e-6f;
  float inv = 1.f / (2.f * bw * bw);
  float xv = x[(size_t)b * ldx + f];
  #pragma unroll
  for (int g = 0; g < GG_; ++g) {
    float gv = -2.f + 4.f * (float)g / (float)(GG_ - 1);
    float d = xv - gv;
    basis[((size_t)b * F + f) * GG_ + g] = __expf(-d * d * inv);
  }
}

// ---- M=8 dense: out[b,n] = act(A1[b]·W1[n] + A2[b]·W2[n] + bias[n]) ----
// Bandwidth-bound (W rows read once); one block per n, 8-way batched reduce.
__global__ void __launch_bounds__(256)
smallm_k(const float* __restrict__ A1, const float* __restrict__ W1, int K1,
         const float* __restrict__ A2, const float* __restrict__ W2, int K2,
         const float* __restrict__ bias, float* __restrict__ out, int ldo, int act) {
  int n = blockIdx.x;
  float acc[B_];
  #pragma unroll
  for (int b = 0; b < B_; ++b) acc[b] = 0.f;
  const float* w1 = W1 + (size_t)n * K1;
  for (int k = threadIdx.x; k < K1; k += 256) {
    float w = w1[k];
    #pragma unroll
    for (int b = 0; b < B_; ++b) acc[b] += A1[(size_t)b * K1 + k] * w;
  }
  if (K2 > 0) {
    const float* w2 = W2 + (size_t)n * K2;
    for (int k = threadIdx.x; k < K2; k += 256) {
      float w = w2[k];
      #pragma unroll
      for (int b = 0; b < B_; ++b) acc[b] += A2[(size_t)b * K2 + k] * w;
    }
  }
  __shared__ float red[256 * B_];
  #pragma unroll
  for (int b = 0; b < B_; ++b) red[threadIdx.x * B_ + b] = acc[b];
  __syncthreads();
  for (int s = 128; s > 0; s >>= 1) {
    if (threadIdx.x < s) {
      #pragma unroll
      for (int b = 0; b < B_; ++b)
        red[threadIdx.x * B_ + b] += red[(threadIdx.x + s) * B_ + b];
    }
    __syncthreads();
  }
  if (threadIdx.x < B_) {
    float v = red[threadIdx.x] + bias[n];
    if (act == 1) v = silu_f(v);
    out[(size_t)threadIdx.x * ldo + n] = v;
  }
}

// ---- heads: out[b,l,c] = fused[b]·heads_w[c,l] + heads_b[c,l] ----
__global__ void __launch_bounds__(128)
heads_k(const float* __restrict__ fused, const float* __restrict__ hw,
        const float* __restrict__ hb, float* __restrict__ out) {
  int n = blockIdx.x;                  // n = c*OUT_LEN + l
  int c = n / OUT_LEN_, l = n % OUT_LEN_;
  const float* w = hw + (size_t)n * DM_;
  float acc[B_];
  #pragma unroll
  for (int b = 0; b < B_; ++b) acc[b] = 0.f;
  for (int k = threadIdx.x; k < DM_; k += 128) {
    float wv = w[k];
    #pragma unroll
    for (int b = 0; b < B_; ++b) acc[b] += fused[b * DM_ + k] * wv;
  }
  __shared__ float red[128 * B_];
  #pragma unroll
  for (int b = 0; b < B_; ++b) red[threadIdx.x * B_ + b] = acc[b];
  __syncthreads();
  for (int s = 64; s > 0; s >>= 1) {
    if (threadIdx.x < s) {
      #pragma unroll
      for (int b = 0; b < B_; ++b)
        red[threadIdx.x * B_ + b] += red[(threadIdx.x + s) * B_ + b];
    }
    __syncthreads();
  }
  if (threadIdx.x < B_)
    out[((size_t)threadIdx.x * OUT_LEN_ + l) * COUT_ + c] = red[threadIdx.x] + hb[n];
}

// ---------------------------------------------------------------------------
extern "C" void kernel_launch(void* const* d_in, const int* in_sizes, int n_in,
                              void* d_out, int out_size, void* d_ws, size_t ws_size,
                              hipStream_t stream) {
  (void)in_sizes; (void)n_in; (void)out_size; (void)ws_size;
  const float* x_prefix   = (const float*)d_in[0];
  const float* y_aux      = (const float*)d_in[1];
  const float* pre_conv_w = (const float*)d_in[2];
  const float* pre_conv_b = (const float*)d_in[3];
  const float* patch_w    = (const float*)d_in[4];
  const float* patch_b    = (const float*)d_in[5];
  const float* dnorm_w[2]  = {(const float*)d_in[6],  (const float*)d_in[14]};
  const float* din_w[2]    = {(const float*)d_in[7],  (const float*)d_in[15]};
  const float* dconv_w[2]  = {(const float*)d_in[8],  (const float*)d_in[16]};
  const float* dconv_b[2]  = {(const float*)d_in[9],  (const float*)d_in[17]};
  const float* dA_log[2]   = {(const float*)d_in[10], (const float*)d_in[18]};
  const float* dD[2]       = {(const float*)d_in[11], (const float*)d_in[19]};
  const float* ddt_bias[2] = {(const float*)d_in[12], (const float*)d_in[20]};
  const float* dout_w[2]   = {(const float*)d_in[13], (const float*)d_in[21]};
  const float* aux_w        = (const float*)d_in[22];
  const float* aux_b        = (const float*)d_in[23];
  const float* kan1_log_bw  = (const float*)d_in[24];
  const float* kan1_base_w  = (const float*)d_in[25];
  const float* kan1_base_b  = (const float*)d_in[26];
  const float* kan1_spline_w= (const float*)d_in[27];
  const float* kan2_log_bw  = (const float*)d_in[28];
  const float* kan2_base_w  = (const float*)d_in[29];
  const float* kan2_base_b  = (const float*)d_in[30];
  const float* kan2_spline_w= (const float*)d_in[31];
  const float* shared_w     = (const float*)d_in[32];
  const float* shared_b     = (const float*)d_in[33];
  const float* heads_w      = (const float*)d_in[34];
  const float* heads_b      = (const float*)d_in[35];
  float* out = (float*)d_out;

  float* ws = (float*)d_ws;
  size_t off = 0;
  auto alloc = [&](size_t n) -> float* { float* p = ws + off; off += n; return p; };
  float* preA   = alloc((size_t)B_ * LP_ * CIN_ * KPRE_);   // 8192 x 320
  float* xc     = alloc((size_t)B_ * LP_ * DM_);            // 8192 x 1024
  float* patA   = alloc((size_t)MROWS_ * DM_ * PATCH_);     // 1024 x 8192
  float* xseq   = alloc((size_t)MROWS_ * DM_);
  float* hbuf0  = alloc((size_t)MROWS_ * DM_);
  float* hbuf1  = alloc((size_t)MROWS_ * DM_);
  float* normed = alloc((size_t)MROWS_ * DM_);
  float* zx     = alloc((size_t)MROWS_ * DIN_);
  float* xbcc   = alloc((size_t)MROWS_ * CONV_DIM_);
  float* dtb    = alloc((size_t)MROWS_ * NH_);
  float* dab    = alloc((size_t)MROWS_ * NH_);
  float* ysb    = alloc((size_t)MROWS_ * DI_);
  float* fused  = alloc((size_t)B_ * 3 * DM_);
  float* basis1 = alloc((size_t)B_ * 3 * DM_ * GG_);
  float* k1out  = alloc((size_t)B_ * KH_);
  float* basis2 = alloc((size_t)B_ * KH_ * GG_);
  float* k2out  = alloc((size_t)B_ * DM_);
  float* shout  = alloc((size_t)B_ * DM_);
  float* hbuf[2] = { hbuf0, hbuf1 };

  // 1) pre-conv (+silu) as WMMA GEMM: M=8192, N=1024, K=320
  { int tot = B_ * LP_ * CIN_ * KPRE_;
    preconv_im2col<<<(tot + 255) / 256, 256, 0, stream>>>(x_prefix, preA); }
  wmma_gemm_f32bf16<<<dim3((B_ * LP_) / 16, (DM_ + 255) / 256), 128, 0, stream>>>(
      preA, pre_conv_w, pre_conv_b, nullptr, xc, B_ * LP_, DM_, CIN_ * KPRE_, 1);

  // 2) patch conv as WMMA GEMM: M=1024, N=1024, K=8192
  { int tot = MROWS_ * DM_ * PATCH_;
    patch_im2col<<<(tot + 255) / 256, 256, 0, stream>>>(xc, patA); }
  wmma_gemm_f32bf16<<<dim3(MROWS_ / 16, (DM_ + 255) / 256), 128, 0, stream>>>(
      patA, patch_w, patch_b, nullptr, xseq, MROWS_, DM_, DM_ * PATCH_, 0);

  // 3) h_f / h_b
  { int tot = MROWS_ * DM_;
    init_dirs<<<(tot + 255) / 256, 256, 0, stream>>>(xseq, hbuf0, hbuf1); }

  // 4) bidirectional Mamba stacks
  for (int dir = 0; dir < 2; ++dir) {
    for (int i = 0; i < NL_; ++i) {
      rmsnorm_k<<<MROWS_, 256, 0, stream>>>(hbuf[dir], dnorm_w[dir] + (size_t)i * DM_, normed);
      // in-proj: M=1024, N=4160, K=1024
      wmma_gemm_f32bf16<<<dim3(MROWS_ / 16, (DIN_ + 255) / 256), 128, 0, stream>>>(
          normed, din_w[dir] + (size_t)i * DIN_ * DM_, nullptr, nullptr, zx,
          MROWS_, DIN_, DM_, 0);
      { int tot = MROWS_ * NH_;
        dtda_k<<<(tot + 255) / 256, 256, 0, stream>>>(
            zx, ddt_bias[dir] + i * NH_, dA_log[dir] + i * NH_, dtb, dab); }
      { int tot = MROWS_ * CONV_DIM_;
        dwconv_k<<<(tot + 255) / 256, 256, 0, stream>>>(
            zx, dconv_w[dir] + (size_t)i * CONV_DIM_ * KCONV_,
            dconv_b[dir] + (size_t)i * CONV_DIM_, xbcc); }
      scan_k<<<B_ * NH_, 64, 0, stream>>>(zx, xbcc, dtb, dab, dD[dir] + i * NH_, ysb);
      // out-proj + residual: M=1024, N=1024, K=2048
      wmma_gemm_f32bf16<<<dim3(MROWS_ / 16, (DM_ + 255) / 256), 128, 0, stream>>>(
          ysb, dout_w[dir] + (size_t)i * DM_ * DI_, nullptr, hbuf[dir], hbuf[dir],
          MROWS_, DM_, DI_, 0);
    }
  }

  // 5) pooled ctx + aux branch -> fused (B, 3*DM)
  { int tot = 2 * B_ * DM_;
    mean_ctx_k<<<(tot + 255) / 256, 256, 0, stream>>>(hbuf0, hbuf1, fused); }
  smallm_k<<<DM_, 256, 0, stream>>>(y_aux, aux_w, AUX_, nullptr, nullptr, 0,
                                    aux_b, fused + 2 * DM_, 3 * DM_, 1);

  // 6) KAN1 (base + 201MB spline, bandwidth-bound)
  { int tot = B_ * 3 * DM_;
    basis_k<<<(tot + 255) / 256, 256, 0, stream>>>(fused, kan1_log_bw, basis1, 3 * DM_, 3 * DM_); }
  smallm_k<<<KH_, 256, 0, stream>>>(fused, kan1_base_w, 3 * DM_,
                                    basis1, kan1_spline_w, 3 * DM_ * GG_,
                                    kan1_base_b, k1out, KH_, 1);

  // 7) KAN2
  { int tot = B_ * KH_;
    basis_k<<<(tot + 255) / 256, 256, 0, stream>>>(k1out, kan2_log_bw, basis2, KH_, KH_); }
  smallm_k<<<DM_, 256, 0, stream>>>(k1out, kan2_base_w, KH_,
                                    basis2, kan2_spline_w, KH_ * GG_,
                                    kan2_base_b, k2out, DM_, 1);

  // 8) shared linear
  smallm_k<<<DM_, 256, 0, stream>>>(k2out, shared_w, DM_, nullptr, nullptr, 0,
                                    shared_b, shout, DM_, 1);

  // 9) heads (134MB weight, bandwidth-bound reduction)
  heads_k<<<COUT_ * OUT_LEN_, 128, 0, stream>>>(shout, heads_w, heads_b, out);
}
